// GCNModel_2_89300960018655
// MI455X (gfx1250) — matrix operand order repaired and checked by reading, not compiled, not run on hardware
//
#include <hip/hip_runtime.h>
#include <hip/hip_bf16.h>

typedef __attribute__((ext_vector_type(16))) __bf16 v16bf;
typedef __attribute__((ext_vector_type(8)))  float  v8f;

#define NN 50000
#define NE 800000
#define H  128

// ---------------- weight f32 -> bf16 (W1 + 3 hidden Ws), done once per call ----
__global__ void k_convert_weights(const float* __restrict__ W1,
                                  const float* __restrict__ Ws,
                                  __bf16* __restrict__ out) {
    int i = blockIdx.x * blockDim.x + threadIdx.x;
    if (i >= 4 * H * H) return;
    float v = (i < H * H) ? W1[i] : Ws[i - H * H];
    out[i] = (__bf16)v;
}

// ---------------- degree / norm ------------------------------------------------
__global__ void k_deg_init(float* __restrict__ deg, float* __restrict__ acc) {
    int i = blockIdx.x * blockDim.x + threadIdx.x;
    if (i < NN) deg[i] = 1.0f;          // self-loop contributes 1 to in-degree
    if (i < 2)  acc[i] = 0.0f;          // zero the pooling accumulator
}

__global__ void k_deg_accum(const int* __restrict__ col, float* __restrict__ deg) {
    int e = blockIdx.x * blockDim.x + threadIdx.x;
    if (e < NE) unsafeAtomicAdd(&deg[col[e]], 1.0f);
}

__global__ void k_rsqrt(float* __restrict__ deg) {
    int i = blockIdx.x * blockDim.x + threadIdx.x;
    if (i < NN) deg[i] = rsqrtf(deg[i]);   // deg >= 1 always
}

// ---------------- XW = X @ W  (bf16 WMMA, f32 accumulate) ----------------------
// Block: 256 threads = 8 waves. Block owns 16 node-rows; wave w owns N-cols [16w,16w+16).
// K = 128 -> 4 chained v_wmma_f32_16x16x32_bf16 per wave.
__global__ __launch_bounds__(256)
void k_gemm_wmma(const float* __restrict__ X, const __bf16* __restrict__ Wb,
                 float* __restrict__ XW) {
    __shared__ __bf16 sW[H][H + 8];
    __shared__ __bf16 sX[16][H + 8];
    const int nodeRow = blockIdx.x * 16;

    for (int idx = threadIdx.x; idx < H * H; idx += 256)
        sW[idx >> 7][idx & 127] = Wb[idx];
    for (int idx = threadIdx.x; idx < 16 * H; idx += 256) {
        int r = idx >> 7, k = idx & 127;
        sX[r][k] = (__bf16)X[(size_t)(nodeRow + r) * H + k];
    }
    __syncthreads();

    const int lane = threadIdx.x & 31;
    const int wid  = threadIdx.x >> 5;   // column tile 0..7
    const int half = lane >> 4;          // lane group
    const int m    = lane & 15;

    v8f c = {};
#pragma unroll
    for (int kb = 0; kb < 4; ++kb) {
        const int kbase = kb * 32;
        v16bf a, b;
#pragma unroll
        for (int v = 0; v < 8; ++v) {
            // A 16x32 bf16 layout: lanes 0-15 hold K{0..7,16..23}, lanes 16-31 +8
            const int kA = kbase + ((v < 4) ? (2 * v) : (16 + 2 * (v - 4))) + half * 8;
            a[2 * v]     = sX[m][kA];
            a[2 * v + 1] = sX[m][kA + 1];
            // B 32x16 bf16 layout: lane holds column N=lane&15, same K order as A
            const int kB = kA;
            b[2 * v]     = sW[kB][wid * 16 + m];
            b[2 * v + 1] = sW[kB + 1][wid * 16 + m];
        }
        c = __builtin_amdgcn_wmma_f32_16x16x32_bf16(
                /*neg_a=*/false, a, /*neg_b=*/false, b,
                /*c_mod=*/(short)0, c, /*reuse_a=*/false, /*reuse_b=*/false);
    }
    // C/D layout: VGPR r -> M = r + 8*half, N = lane&15
#pragma unroll
    for (int r = 0; r < 8; ++r)
        XW[(size_t)(nodeRow + r + 8 * half) * H + wid * 16 + m] = c[r];
}

// ---------------- out = bias + xw[i]*dinv[i]^2  (self-loop term) ---------------
// One wave handles one node (32 lanes x float4 = 128 features); node index is
// forced into an SGPR so dinv[n] becomes a scalar (s_load) access.
__global__ void k_self_bias(const float* __restrict__ XW, const float* __restrict__ dinv,
                            const float* __restrict__ bias, float* __restrict__ out) {
    int t = blockIdx.x * blockDim.x + threadIdx.x;     // NN * 32 float4 slots
    int n = __builtin_amdgcn_readfirstlane(t >> 5);    // wave-uniform node id
    if (n >= NN) return;
    int seg = threadIdx.x & 31;
    float d = dinv[n];                                 // uniform -> scalar load
    float w = d * d;
    float4 xv = ((const float4*)XW)[(size_t)n * 32 + seg];
    float4 bv = ((const float4*)bias)[seg];
    float4 o;
    o.x = bv.x + xv.x * w;  o.y = bv.y + xv.y * w;
    o.z = bv.z + xv.z * w;  o.w = bv.w + xv.w * w;
    ((float4*)out)[(size_t)n * 32 + seg] = o;
}

// ---------------- edge scatter: out[col] += xw[row]*dinv[row]*dinv[col] --------
// One wave per edge (32 lanes x float4 = 128 features). Edge id is wave-uniform
// and forced into an SGPR: row/col/dinv loads become scalar (s_load, KMcnt) and
// only the 128-bit gather + 4 non-returning L2 f32 atomics stay on the vector
// path. xw/out are L2-resident (25.6 MB each, vs 192 MB L2).
__global__ void k_edge_scatter(const int* __restrict__ row, const int* __restrict__ col,
                               const float* __restrict__ dinv, const float* __restrict__ XW,
                               float* __restrict__ out) {
    long long t = (long long)blockIdx.x * blockDim.x + threadIdx.x;
    int e = __builtin_amdgcn_readfirstlane((int)(t >> 5));   // wave-uniform edge id
    if (e >= NE) return;
    int seg = threadIdx.x & 31;
    int r = row[e];                    // uniform -> s_load
    int c = col[e];                    // uniform -> s_load
    float w = dinv[r] * dinv[c];       // uniform -> s_load + broadcast mul
    float4 xv = ((const float4*)XW)[(size_t)r * 32 + seg];   // coalesced 512B/wave
    float* dst = out + (size_t)c * H + seg * 4;
    unsafeAtomicAdd(dst + 0, xv.x * w);
    unsafeAtomicAdd(dst + 1, xv.y * w);
    unsafeAtomicAdd(dst + 2, xv.z * w);
    unsafeAtomicAdd(dst + 3, xv.w * w);
}

// ---------------- activations --------------------------------------------------
__global__ void k_relu(float* __restrict__ x) {
    int t = blockIdx.x * blockDim.x + threadIdx.x;
    if (t >= NN * (H / 4)) return;
    float4 v = ((float4*)x)[t];
    v.x = fmaxf(v.x, 0.f); v.y = fmaxf(v.y, 0.f);
    v.z = fmaxf(v.z, 0.f); v.w = fmaxf(v.w, 0.f);
    ((float4*)x)[t] = v;
}

__global__ void k_relu_res(float* __restrict__ agg, const float* __restrict__ res) {
    int t = blockIdx.x * blockDim.x + threadIdx.x;
    if (t >= NN * (H / 4)) return;
    float4 a = ((float4*)agg)[t];
    float4 r = ((const float4*)res)[t];
    a.x = fmaxf(a.x + r.x, 0.f); a.y = fmaxf(a.y + r.y, 0.f);
    a.z = fmaxf(a.z + r.z, 0.f); a.w = fmaxf(a.w + r.w, 0.f);
    ((float4*)agg)[t] = a;
}

// ---------------- final FC [128x2] + mean pool ---------------------------------
__global__ __launch_bounds__(256)
void k_fc_pool(const float* __restrict__ x, const float* __restrict__ Wfc,
               float* __restrict__ acc) {
    __shared__ float s0[256], s1[256];
    float a0 = 0.f, a1 = 0.f;
    for (int n = blockIdx.x * blockDim.x + threadIdx.x; n < NN;
         n += gridDim.x * blockDim.x) {
        const float* xr = x + (size_t)n * H;
        float d0 = 0.f, d1 = 0.f;
#pragma unroll 8
        for (int k = 0; k < H; ++k) {
            float xv = xr[k];
            d0 += xv * Wfc[2 * k];      // uniform address -> scalar cache
            d1 += xv * Wfc[2 * k + 1];
        }
        a0 += d0; a1 += d1;
    }
    s0[threadIdx.x] = a0; s1[threadIdx.x] = a1;
    __syncthreads();
    for (int off = 128; off > 0; off >>= 1) {
        if (threadIdx.x < off) {
            s0[threadIdx.x] += s0[threadIdx.x + off];
            s1[threadIdx.x] += s1[threadIdx.x + off];
        }
        __syncthreads();
    }
    if (threadIdx.x == 0) {
        unsafeAtomicAdd(&acc[0], s0[0]);
        unsafeAtomicAdd(&acc[1], s1[0]);
    }
}

__global__ void k_finalize(const float* __restrict__ acc, const float* __restrict__ bfc,
                           float* __restrict__ out) {
    int j = threadIdx.x;
    if (j < 2) out[j] = acc[j] * (1.0f / (float)NN) + bfc[j];
}

// ==============================================================================
extern "C" void kernel_launch(void* const* d_in, const int* in_sizes, int n_in,
                              void* d_out, int out_size, void* d_ws, size_t ws_size,
                              hipStream_t stream) {
    const float* node = (const float*)d_in[0];
    const int*   eidx = (const int*)d_in[1];           // [2, NE] int
    const float* W1   = (const float*)d_in[3];
    const float* b1   = (const float*)d_in[4];
    const float* Ws   = (const float*)d_in[5];         // [3,128,128]
    const float* bs   = (const float*)d_in[6];         // [3,128]
    const float* Wfc  = (const float*)d_in[7];
    const float* bfc  = (const float*)d_in[8];
    float* out = (float*)d_out;

    const int* erow = eidx;
    const int* ecol = eidx + NE;

    // ---- workspace carve (all offsets 256B-aligned) ----
    char* ws = (char*)d_ws;
    size_t o = 0;
    float*  dinv = (float*)(ws + o);  o += 50048u * 4u;          // 200192
    float*  acc  = (float*)(ws + o);  o += 256u;
    __bf16* wbf  = (__bf16*)(ws + o); o += 4u * H * H * 2u;      // 131072
    float*  XW   = (float*)(ws + o);  o += (size_t)NN * H * 4u;  // 25.6 MB
    float*  bufA = (float*)(ws + o);  o += (size_t)NN * H * 4u;
    float*  bufB = (float*)(ws + o);  o += (size_t)NN * H * 4u;

    const int ROWT  = NN / 16;                    // 3125 exact
    const int NH4   = NN * (H / 4);               // elementwise float4 threads
    const int BLK   = 256;
    const int gElem = (NH4 + BLK - 1) / BLK;      // 6250
    const int gEdge = (int)(((long long)NE * 32 + BLK - 1) / BLK);  // 100000

    // norm precompute
    k_convert_weights<<<(4 * H * H + BLK - 1) / BLK, BLK, 0, stream>>>(W1, Ws, wbf);
    k_deg_init<<<(NN + BLK - 1) / BLK, BLK, 0, stream>>>(dinv, acc);
    k_deg_accum<<<(NE + BLK - 1) / BLK, BLK, 0, stream>>>(ecol, dinv);
    k_rsqrt<<<(NN + BLK - 1) / BLK, BLK, 0, stream>>>(dinv);

    // ---- layer 0: x1 = relu(gcn(node, W1, b1)) -> bufA ----
    k_gemm_wmma<<<ROWT, BLK, 0, stream>>>(node, wbf + 0 * H * H, XW);
    k_self_bias<<<gElem, BLK, 0, stream>>>(XW, dinv, b1, bufA);
    k_edge_scatter<<<gEdge, BLK, 0, stream>>>(erow, ecol, dinv, XW, bufA);
    k_relu<<<gElem, BLK, 0, stream>>>(bufA);

    // ---- layers 1..3: x = relu(gcn(x, Ws[i], bs[i]) + x), ping-pong A/B ----
    float* xin  = bufA;
    float* xout = bufB;
    for (int i = 0; i < 3; ++i) {
        k_gemm_wmma<<<ROWT, BLK, 0, stream>>>(xin, wbf + (size_t)(i + 1) * H * H, XW);
        k_self_bias<<<gElem, BLK, 0, stream>>>(XW, dinv, bs + (size_t)i * H, xout);
        k_edge_scatter<<<gEdge, BLK, 0, stream>>>(erow, ecol, dinv, XW, xout);
        k_relu_res<<<gElem, BLK, 0, stream>>>(xout, xin);
        float* t = xin; xin = xout; xout = t;     // result now in xin
    }

    // ---- final FC + global mean pool ----
    k_fc_pool<<<256, BLK, 0, stream>>>(xin, Wfc, acc);
    k_finalize<<<1, 32, 0, stream>>>(acc, bfc, out);

    (void)in_sizes; (void)n_in; (void)out_size; (void)ws_size; (void)d_in;
}
